// DisentangleLoss_16252156248113
// MI455X (gfx1250) — matrix operand order repaired and checked by reading, not compiled, not run on hardware
//
#include <hip/hip_runtime.h>

typedef __attribute__((ext_vector_type(16))) _Float16 v16h;
typedef __attribute__((ext_vector_type(8)))  _Float16 v8h;
typedef __attribute__((ext_vector_type(4)))  _Float16 v4h;
typedef __attribute__((ext_vector_type(8)))  float    v8f;
typedef __attribute__((ext_vector_type(4)))  float    v4f;

#define KROWS   9216      // N*BS
#define DDIM    128
#define NBLK    9         // N (views)
#define NT      576       // KROWS/16 column tiles
#define EPSN    1e-8f
// Fold BOTH 1/temp=2 and log2(e) into sn:
// (c*u_i)·(c*u_j) = 2*log2e*cos = logit*log2e, so 2^acc = exp(logit).
#define NSCALE  1.6986436e+0f   // sqrt(2 * log2(e)) = sqrt(2.88539008)
#define LN2F    0.6931471805599453f

#if __has_builtin(__builtin_amdgcn_exp2f)
#define EXP2F(x) __builtin_amdgcn_exp2f(x)
#else
#define EXP2F(x) exp2f(x)
#endif

union h16u { v16h v; v8h h[2]; };

__global__ void dl_zero_out(float* out) { out[0] = 0.0f; }

// One wave (32 lanes) per row: fp32 norm, write sqrt(2*log2e)-scaled row as f16.
__global__ __launch_bounds__(256) void dl_normalize(const float* __restrict__ s,
                                                    _Float16* __restrict__ sn) {
  const int lane = threadIdx.x & 31;
  const int row  = blockIdx.x * 8 + (threadIdx.x >> 5);
  const float* rp = s + (size_t)row * DDIM;
  v4f x = ((const v4f*)rp)[lane];
  float ss = x.x * x.x + x.y * x.y + x.z * x.z + x.w * x.w;
#pragma unroll
  for (int off = 16; off >= 1; off >>= 1) ss += __shfl_xor(ss, off, 32);
  const float scale = NSCALE / fmaxf(sqrtf(ss), EPSN);
  v4h o;
  o.x = (_Float16)(x.x * scale);
  o.y = (_Float16)(x.y * scale);
  o.z = (_Float16)(x.z * scale);
  o.w = (_Float16)(x.w * scale);
  ((v4h*)(sn + (size_t)row * DDIM))[lane] = o;
}

// B fragment for one 16-col tile: lane holds col n=lane&15, K-half by lane>=16.
__device__ __forceinline__ void load_btile(const _Float16* __restrict__ sn,
                                           int ct, int m, int hi, v16h* bb) {
  const _Float16* brow = sn + (size_t)(ct * 16 + m) * DDIM + (hi << 4);
#pragma unroll
  for (int c = 0; c < 4; ++c) bb[c] = *(const v16h*)(brow + c * 32);
}

// A fragments for one 16-row tile (contiguous k = 16*hi + s, same order as B), 4 K-chunks.
__device__ __forceinline__ void load_atile(const _Float16* __restrict__ sn,
                                           int rt, int m, int hi8, v16h* a) {
  const _Float16* arow = sn + (size_t)(rt * 16 + m) * DDIM;
#pragma unroll
  for (int c = 0; c < 4; ++c) {
    h16u u;
    u.h[0] = *(const v8h*)(arow + c * 32 + 2 * hi8);       // K = KB + 16*hi + [0,8)
    u.h[1] = *(const v8h*)(arow + c * 32 + 2 * hi8 + 8);  // K = KB + 16*hi + 8 + [0,8)
    a[c] = u.v;
  }
}

// One wave per PAIR of 16-row tiles: each B fragment feeds 8 WMMAs,
// B loads double-buffered so L2 latency hides under WMMA + exp work.
// Accumulator is logit*log2e, so sum-exp is a bare v_exp_f32 per element.
__global__ __launch_bounds__(256) void dl_loss(const _Float16* __restrict__ sn,
                                               float* __restrict__ out) {
  const int lane = threadIdx.x & 31;
  const int wid  = blockIdx.x * 8 + (threadIdx.x >> 5);  // 0..287
  const int rt0  = wid * 2;
  const int rt1  = rt0 + 1;
  const int m    = lane & 15;
  const int hi   = lane >> 4;
  const int hi8  = hi * 8;
  const int selfb = rt0 >> 6;            // shared view-block for both row tiles

  v16h a0[4], a1[4];
  load_atile(sn, rt0, m, hi8, a0);
  load_atile(sn, rt1, m, hi8, a1);

  float s0[8] = {}, s1[8] = {};
  float posv0[NBLK] = {}, posv1[NBLK] = {};

  v16h cur[4], nxt[4];
  load_btile(sn, 0, m, hi, cur);

#pragma unroll 2
  for (int ct = 0; ct < NT; ++ct) {
    const int nct = (ct + 1 < NT) ? ct + 1 : ct;
    load_btile(sn, nct, m, hi, nxt);     // overlaps with WMMA + exp below

    v8f acc0 = {}, acc1 = {};
#pragma unroll
    for (int c = 0; c < 4; ++c)
      acc0 = __builtin_amdgcn_wmma_f32_16x16x32_f16(
          false, a0[c], false, cur[c], (short)0, acc0, false, false);
#pragma unroll
    for (int c = 0; c < 4; ++c)
      acc1 = __builtin_amdgcn_wmma_f32_16x16x32_f16(
          false, a1[c], false, cur[c], (short)0, acc1, false, false);

    const bool m0 = (((ct ^ rt0) & 63) == 0);   // wave-uniform
    const bool m1 = (((ct ^ rt1) & 63) == 0);   // wave-uniform, exclusive with m0
    if (__builtin_expect(m0 | m1, 0)) {
      // Rare path: tile contains the mod-1024 diagonal (self or positive).
      const int  b     = ct >> 6;
      const bool self0 = (ct == rt0);
      const bool self1 = (ct == rt1);
#pragma unroll
      for (int v = 0; v < 8; ++v) {
        const float x0 = acc0[v], x1 = acc1[v];
        const bool d0 = m0 && ((v + hi8) == m);
        const bool d1 = m1 && ((v + hi8) == m);
        if (d0) { if (!self0) posv0[b] = x0; } else s0[v] += EXP2F(x0);
        if (d1) { if (!self1) posv1[b] = x1; } else s1[v] += EXP2F(x1);
      }
    } else {
      // Fast path: all 256+256 elements are negatives -> pure exp2+add.
#pragma unroll
      for (int v = 0; v < 8; ++v) {
        s0[v] += EXP2F(acc0[v]);
        s1[v] += EXP2F(acc1[v]);
      }
    }

#pragma unroll
    for (int c = 0; c < 4; ++c) cur[c] = nxt[c];
  }

  // Reduce per-row sums across the 16 lanes of each half-wave.
#pragma unroll
  for (int v = 0; v < 8; ++v) {
#pragma unroll
    for (int off = 1; off <= 8; off <<= 1) {
      s0[v] += __shfl_xor(s0[v], off, 32);
      s1[v] += __shfl_xor(s1[v], off, 32);
    }
  }

  // Owner lanes (0..7 rows 0-7, 24..31 rows 8-15) finish their row.
  // posv holds p' = logit*log2e: exp(logit)=2^p', logit = p'*ln2.
  float local = 0.0f;
  if ((lane < 8) || (lane >= 24)) {
    const int vsel = lane & 7;
    float S0 = 0.0f, S1 = 0.0f;
#pragma unroll
    for (int v = 0; v < 8; ++v)
      if (v == vsel) { S0 = s0[v]; S1 = s1[v]; }
#pragma unroll
    for (int b = 0; b < NBLK; ++b)
      if (b != selfb) {
        const float p0 = posv0[b];
        const float p1 = posv1[b];
        local += __logf(EXP2F(p0) + S0) - p0 * LN2F;  // logaddexp(p, log S) - p
        local += __logf(EXP2F(p1) + S1) - p1 * LN2F;
      }
  }
#pragma unroll
  for (int off = 16; off >= 1; off >>= 1) local += __shfl_xor(local, off, 32);
  if (lane == 0)
    atomicAdd(out, local * (1.0f / ((float)KROWS * (float)(NBLK - 1))));
}

extern "C" void kernel_launch(void* const* d_in, const int* in_sizes, int n_in,
                              void* d_out, int out_size, void* d_ws, size_t ws_size,
                              hipStream_t stream) {
  const float* s   = (const float*)d_in[0];
  float*       out = (float*)d_out;
  _Float16*    sn  = (_Float16*)d_ws;   // K*D f16 = 2.36 MB of scratch

  dl_zero_out<<<dim3(1), dim3(1), 0, stream>>>(out);
  dl_normalize<<<dim3(KROWS / 8), dim3(256), 0, stream>>>(s, sn);
  dl_loss<<<dim3(NT / 16), dim3(256), 0, stream>>>(sn, out);
}
